// WaveletChebConv_51960514347290
// MI455X (gfx1250) — matrix operand order repaired and checked by reading, not compiled, not run on hardware
//
#include <hip/hip_runtime.h>
#include <cmath>

#define NN 8192          // nodes
#define ND 128           // in/out dim
#define SPLIT 4          // K-split factor for the big GEMMs
#define KSEG (NN / SPLIT)   // 2048 K per block
#define KBS (KSEG / 32)     // 64 k-steps per block

typedef __bf16 bf16_t;
typedef __attribute__((ext_vector_type(16))) __bf16 v16bf;
typedef __attribute__((ext_vector_type(8)))  float  v8f;

union V16 {
  v16bf v;
  float4 f4[2];
  __bf16 h[16];
};

// Build A fragment (16x32 bf16, ISA layout): lane l holds row (l&15),
// K = 8*(l>>4)+{0..7} in halves 0..7 and +16+{0..7} in halves 8..15.
// p points at (row, kbase + 8*(l>>4)) in row-major f32. Scales by s.
__device__ __forceinline__ v16bf make_a_frag(const float* __restrict__ p, float s) {
  float4 a0 = *(const float4*)(p + 0);
  float4 a1 = *(const float4*)(p + 4);
  float4 a2 = *(const float4*)(p + 16);
  float4 a3 = *(const float4*)(p + 20);
  V16 r;
  r.h[ 0] = (__bf16)(a0.x * s); r.h[ 1] = (__bf16)(a0.y * s);
  r.h[ 2] = (__bf16)(a0.z * s); r.h[ 3] = (__bf16)(a0.w * s);
  r.h[ 4] = (__bf16)(a1.x * s); r.h[ 5] = (__bf16)(a1.y * s);
  r.h[ 6] = (__bf16)(a1.z * s); r.h[ 7] = (__bf16)(a1.w * s);
  r.h[ 8] = (__bf16)(a2.x * s); r.h[ 9] = (__bf16)(a2.y * s);
  r.h[10] = (__bf16)(a2.z * s); r.h[11] = (__bf16)(a2.w * s);
  r.h[12] = (__bf16)(a3.x * s); r.h[13] = (__bf16)(a3.y * s);
  r.h[14] = (__bf16)(a3.z * s); r.h[15] = (__bf16)(a3.w * s);
  return r.v;
}

// B fragment (32x16 bf16): lane l holds column (l&15), K = 8*(l>>4)+{0..7} and 16+8*(l>>4)+{0..7}.
// The B buffers are stored pre-swizzled so this is two 16-byte chunks.
__device__ __forceinline__ v16bf load_b_frag(const bf16_t* __restrict__ p) {
  V16 r;
  r.f4[0] = *(const float4*)(p + 0);
  r.f4[1] = *(const float4*)(p + 16);
  return r.v;
}

// Swizzled B layout: elem (t, n) -> [(t>>5)*4096 + n*32 + (t&31)]  (bf16)
__global__ __launch_bounds__(256)
void prep_B(const float* __restrict__ x, bf16_t* __restrict__ B0) {
  int idx = blockIdx.x * blockDim.x + threadIdx.x;   // 0 .. NN*ND-1
  int t = idx >> 7, n = idx & 127;
  B0[(t >> 5) * 4096 + n * 32 + (t & 31)] = (bf16_t)x[idx];
}

// theta (k,i,o) f32 -> swizzled bf16: [k*16384 + (i>>5)*4096 + o*32 + (i&31)]
__global__ __launch_bounds__(256)
void prep_theta(const float* __restrict__ th, bf16_t* __restrict__ thS) {
  int idx = blockIdx.x * blockDim.x + threadIdx.x;   // 0 .. 5*128*128-1
  int k = idx / 16384, rem = idx & 16383;
  int i = rem >> 7, o = rem & 127;
  thS[k * 16384 + (i >> 5) * 4096 + o * 32 + (i & 31)] = (bf16_t)th[idx];
}

// Partial GEMM: Cpart[ks] = L[:, ks*KSEG:(ks+1)*KSEG] @ B[ks*KSEG:...]
// Block: 64 threads = 2 waves; wave computes 16 rows x 128 cols over one
// K segment. blockIdx: low 8 bits = row block, high bits = K split.
__global__ __launch_bounds__(64)
__attribute__((amdgpu_waves_per_eu(2, 4)))
void cheb_gemm(const float* __restrict__ Lm, const bf16_t* __restrict__ Bin,
               float* __restrict__ Cpart) {
  const int lane = threadIdx.x & 31;
  const int wave = threadIdx.x >> 5;
  const int ln = lane & 15;
  const int sel = lane >> 4;
  const int rb = blockIdx.x & 255;
  const int ks = blockIdx.x >> 8;
  const int rowBase = rb * 32 + wave * 16;
  const int rowA = rowBase + ln;
  const int c0 = sel * 8;

  v8f acc[8];
  #pragma unroll
  for (int i = 0; i < 8; ++i) acc[i] = (v8f){};

  const float* arow = Lm + (size_t)rowA * NN + ks * KSEG + c0;
  const bf16_t* bwave = Bin + ks * (KBS * 4096) + ln * 32 + sel * 8;

  for (int kb = 0; kb < KBS; ++kb) {
    // keep the HBM stream of L deep: prefetch ~3KB ahead, wrapped in-segment
    __builtin_prefetch(arow + ((kb + 24) & (KBS - 1)) * 32, 0, 3);

    v16bf afrag = make_a_frag(arow + kb * 32, 1.0f);

    // Preload all 8 B fragments (56 extra VGPRs allowed by waves_per_eu hint)
    // so the compiler emits one 16-load clause + partial waits, then 8
    // back-to-back WMMAs that overlap the load tail.
    const bf16_t* bbase = bwave + kb * 4096;
    v16bf bfrag[8];
    #pragma unroll
    for (int nt = 0; nt < 8; ++nt) bfrag[nt] = load_b_frag(bbase + nt * 512);

    #pragma unroll
    for (int nt = 0; nt < 8; ++nt)
      acc[nt] = __builtin_amdgcn_wmma_f32_16x16x32_bf16(
          false, afrag, false, bfrag[nt], (short)0, acc[nt], false, false);
  }

  // Write partial sums. C/D layout: VGPR v, lane l: row = v + 8*(l>>4), col = l&15.
  float* cp = Cpart + (size_t)ks * (NN * ND);
  #pragma unroll
  for (int nt = 0; nt < 8; ++nt) {
    int col = nt * 16 + ln;
    #pragma unroll
    for (int v = 0; v < 8; ++v) {
      int row = rowBase + sel * 8 + v;
      cp[(size_t)row * ND + col] = acc[nt][v];
    }
  }
}

// Tout = alpha * sum_s Cpart[s] - (useprev ? Tprev : 0); Bout = swizzle(bf16(Tout))
__global__ __launch_bounds__(256)
void cheb_reduce(const float* __restrict__ Cpart, const float* __restrict__ Tprev,
                 float alpha, int useprev,
                 float* __restrict__ Tout, bf16_t* __restrict__ Bout) {
  int idx = blockIdx.x * blockDim.x + threadIdx.x;   // 0 .. NN*ND-1
  float s = Cpart[idx];
  #pragma unroll
  for (int p = 1; p < SPLIT; ++p) s += Cpart[(size_t)p * (NN * ND) + idx];
  float cv = alpha * s;
  if (useprev) cv -= Tprev[idx];
  Tout[idx] = cv;
  int t = idx >> 7, n = idx & 127;
  Bout[(t >> 5) * 4096 + n * 32 + (t & 31)] = (bf16_t)cv;
}

// out = sum_k c_k * T_k @ theta_k   (T_0 = x); theta pre-swizzled bf16.
__global__ __launch_bounds__(64)
__attribute__((amdgpu_waves_per_eu(2, 4)))
void cheb_final(const float* __restrict__ T0, const float* __restrict__ T1,
                const float* __restrict__ T2, const float* __restrict__ T3,
                const float* __restrict__ T4, const bf16_t* __restrict__ thS,
                float cc0, float cc1, float cc2, float cc3, float cc4,
                float* __restrict__ out) {
  const int lane = threadIdx.x & 31;
  const int wave = threadIdx.x >> 5;
  const int ln = lane & 15;
  const int sel = lane >> 4;
  const int rowBase = blockIdx.x * 32 + wave * 16;
  const int rowA = rowBase + ln;
  const int c0 = sel * 8;

  const float* Ts[5] = {T0, T1, T2, T3, T4};
  const float cs[5] = {cc0, cc1, cc2, cc3, cc4};

  v8f acc[8];
  #pragma unroll
  for (int i = 0; i < 8; ++i) acc[i] = (v8f){};

  #pragma unroll
  for (int k = 0; k < 5; ++k) {
    const float* ap = Ts[k] + (size_t)rowA * ND + c0;
    const bf16_t* bwave = thS + k * 16384 + ln * 32 + sel * 8;
    #pragma unroll
    for (int kb = 0; kb < ND / 32; ++kb) {
      v16bf afrag = make_a_frag(ap + kb * 32, cs[k]);  // fold coeff into A
      const bf16_t* bbase = bwave + kb * 4096;
      v16bf bfrag[8];
      #pragma unroll
      for (int nt = 0; nt < 8; ++nt) bfrag[nt] = load_b_frag(bbase + nt * 512);
      #pragma unroll
      for (int nt = 0; nt < 8; ++nt)
        acc[nt] = __builtin_amdgcn_wmma_f32_16x16x32_bf16(
            false, afrag, false, bfrag[nt], (short)0, acc[nt], false, false);
    }
  }

  #pragma unroll
  for (int nt = 0; nt < 8; ++nt) {
    int col = nt * 16 + ln;
    #pragma unroll
    for (int v = 0; v < 8; ++v) {
      int row = rowBase + sel * 8 + v;
      out[(size_t)row * ND + col] = acc[nt][v];
    }
  }
}

extern "C" void kernel_launch(void* const* d_in, const int* in_sizes, int n_in,
                              void* d_out, int out_size, void* d_ws, size_t ws_size,
                              hipStream_t stream) {
  (void)in_sizes; (void)n_in; (void)out_size; (void)ws_size;
  const float* x  = (const float*)d_in[0];   // (8192,128)
  const float* Lm = (const float*)d_in[1];   // (8192,8192)
  const float* th = (const float*)d_in[2];   // (5,128,128)
  float* out = (float*)d_out;                // (8192,128)

  char* ws = (char*)d_ws;
  const size_t MB = 1024 * 1024;
  bf16_t* B0  = (bf16_t*)(ws + 0 * MB);    // 2 MB swizzled bf16 T_cur
  bf16_t* B1  = (bf16_t*)(ws + 2 * MB);    // 2 MB ping-pong
  float*  T1  = (float*)(ws + 4 * MB);     // 4 MB each
  float*  T2  = (float*)(ws + 8 * MB);
  float*  T3  = (float*)(ws + 12 * MB);
  float*  T4  = (float*)(ws + 16 * MB);
  bf16_t* thS = (bf16_t*)(ws + 20 * MB);   // 160 KB swizzled theta
  float*  Cp  = (float*)(ws + 21 * MB);    // SPLIT x 4 MB partial C buffers

  // c_k = (2/K) * exp(-tau * cos(pi*(k+0.5)/K)), K=5, tau=0.5
  float cs[5];
  for (int k = 0; k < 5; ++k) {
    float xk = cosf(3.14159265358979323846f * (k + 0.5f) / 5.0f);
    cs[k] = (2.0f / 5.0f) * expf(-0.5f * xk);
  }

  prep_B<<<(NN * ND) / 256, 256, 0, stream>>>(x, B0);
  prep_theta<<<(5 * ND * ND) / 256, 256, 0, stream>>>(th, thS);

  const int gemmGrid = 256 * SPLIT;          // row blocks x K splits
  const int redGrid  = (NN * ND) / 256;

  // T1 = L@x ; Tk = 2*(L@T_{k-1}) - T_{k-2}
  cheb_gemm<<<gemmGrid, 64, 0, stream>>>(Lm, B0, Cp);
  cheb_reduce<<<redGrid, 256, 0, stream>>>(Cp, x,  1.0f, 0, T1, B1);
  cheb_gemm<<<gemmGrid, 64, 0, stream>>>(Lm, B1, Cp);
  cheb_reduce<<<redGrid, 256, 0, stream>>>(Cp, x,  2.0f, 1, T2, B0);
  cheb_gemm<<<gemmGrid, 64, 0, stream>>>(Lm, B0, Cp);
  cheb_reduce<<<redGrid, 256, 0, stream>>>(Cp, T1, 2.0f, 1, T3, B1);
  cheb_gemm<<<gemmGrid, 64, 0, stream>>>(Lm, B1, Cp);
  cheb_reduce<<<redGrid, 256, 0, stream>>>(Cp, T2, 2.0f, 1, T4, B0);

  cheb_final<<<NN / 32, 64, 0, stream>>>(x, T1, T2, T3, T4, thS,
                                         cs[0], cs[1], cs[2], cs[3], cs[4], out);
}
